// BiMultiHeadAttention_15066745274735
// MI455X (gfx1250) — matrix operand order repaired and checked by reading, not compiled, not run on hardware
//
#include <hip/hip_runtime.h>
#include <hip/hip_bf16.h>
#include <math.h>

// ---------------------------------------------------------------------------
// BiMultiHeadAttention for MI455X (gfx1250), bf16 WMMA pipeline.
// B=8 T=2048 S=512 VD=512 AD=256 E=512 H=8 D=64.
// Compute-bound (~55 GFLOP vs ~75MB traffic) -> all GEMMs through
// v_wmma_f32_16x16x32_bf16. attn_a == attn_v^T (both softmax over T), so one
// bf16 score tensor [B,H,T,S] is softmaxed once and consumed twice.
// This round: NT template (128x128 block tile / 8 WMMAs per wave per k-step
// for the wide-N GEMMs, 128x64 for N=64), and fragment loads split from WMMA
// issue so the 4-8 WMMAs run back-to-back after a single s_wait_dscnt.
// ---------------------------------------------------------------------------

typedef __attribute__((ext_vector_type(16))) __bf16 v16bf;
typedef __attribute__((ext_vector_type(8)))  float  v8f;
typedef __attribute__((ext_vector_type(8)))  unsigned short us8;

#ifndef CLAMP_VAL
#define CLAMP_VAL 50000.0f
#endif

__device__ __forceinline__ float bf2f(unsigned short u) {
    union { unsigned int i; float f; } c;
    c.i = ((unsigned int)u) << 16;
    return c.f;
}
__device__ __forceinline__ unsigned short f2bf(float f) {
    union { float f; unsigned int i; } c;
    c.f = f;
    unsigned int u = c.i + 0x7FFFu + ((c.i >> 16) & 1u); // round-to-nearest-even
    return (unsigned short)(u >> 16);
}

// ---------------------------------------------------------------------------
// f32 -> bf16 elementwise convert (grid-stride)
// ---------------------------------------------------------------------------
__global__ void k_f32_to_bf16(const float* __restrict__ in,
                              unsigned short* __restrict__ out, int n) {
    int i = blockIdx.x * blockDim.x + threadIdx.x;
    int stride = gridDim.x * blockDim.x;
    for (; i < n; i += stride) out[i] = f2bf(in[i]);
}

// ---------------------------------------------------------------------------
// Generic stride-parameterized bf16 WMMA GEMM:
//   C[b0,b1,m,n] = clamp( (sum_k A[m,k]*B[k,n] + bias[n]) * scale )
// Block: 256 threads (8 wave32). Block tile 128(M) x NT*16(N), K-step 32.
// Wave w computes rows [w*16, w*16+16) x all NT*16 cols = NT WMMAs/k-step.
// AMODE: 0 = A unit-stride along k, 1 = unit-stride along m.
// BMODE: 0 = B unit-stride along k, 1 = unit-stride along n.
// NT: 4 (N tile 64) or 8 (N tile 128).
// A tile LDS: row-major [128][32]; B tile LDS: transposed [n][k] so fragment
// columns are two contiguous 16B runs (-> ds_load_b128 pairs).
// Double-buffered LDS + register staging: one barrier per k-step; all
// fragments loaded before the WMMA burst (single s_wait_dscnt).
// ---------------------------------------------------------------------------
template <bool OUT_BF16, int AMODE, int BMODE, int NT>
__global__ __launch_bounds__(256) void k_gemm_bf16_wmma(
    const unsigned short* __restrict__ A, long long ldaM, long long ldaK,
    long long ldaB0, long long ldaB1,
    const unsigned short* __restrict__ Bm, long long ldbK, long long ldbN,
    long long ldbB0, long long ldbB1,
    void* __restrict__ Cv, long long ldcM, long long ldcN,
    long long ldcB0, long long ldcB1,
    const float* __restrict__ bias, float scale, float clampv,
    int K, int nH) {
    __shared__ unsigned short As[2][128][40];     // [m][k], 80B rows
    __shared__ unsigned short Bt[2][NT * 16][40]; // [n][k] transposed

    const int tid  = threadIdx.x;
    const int lane = tid & 31;
    const int wave = tid >> 5;   // 0..7 -> M sub-tile
    const int hi   = lane >> 4;  // lane half (ISA fragment layout)
    const int l16  = lane & 15;

    const int z  = blockIdx.z;
    const int b0 = z / nH;
    const int b1 = z - b0 * nH;
    A  += (long long)b0 * ldaB0 + (long long)b1 * ldaB1;
    Bm += (long long)b0 * ldbB0 + (long long)b1 * ldbB1;
    const long long cbase = (long long)b0 * ldcB0 + (long long)b1 * ldcB1;

    const int m0 = blockIdx.y * 128;
    const int n0 = blockIdx.x * (NT * 16);

    v8f acc[NT] = {};

    // ---- per-thread tile-load setup (compile-time mode selection) ----
    int a_r0, a_c0;
    const unsigned short* pA;
    long long aStep, aSecond;
    if constexpr (AMODE == 0) {       // 2 x b128 along k
        a_c0 = (tid & 3) * 8;         // k sub-run
        a_r0 = tid >> 2;              // rows a_r0 and a_r0+64
        pA = A + (long long)(m0 + a_r0) * ldaM + a_c0;
        aStep = 32;
        aSecond = (long long)64 * ldaM;
    } else {                          // 2 x b128 along m, scatter to 16 rows
        a_c0 = tid >> 3;              // k column 0..31
        a_r0 = (tid & 7) * 16;        // row group base
        pA = A + (long long)(m0 + a_r0) + (long long)a_c0 * ldaK;
        aStep = 32 * ldaK;
        aSecond = 8;
    }

    int b_n0, b_k0;
    const unsigned short* pB;
    long long bStep, bSecond;
    if constexpr (BMODE == 0) {       // b128 along k -> contiguous Bt row
        b_k0 = (tid & 3) * 8;
        b_n0 = tid >> 2;              // 0..63 (NT=8: also +64)
        pB = Bm + b_k0 + (long long)(n0 + b_n0) * ldbN;
        bStep = 32;
        bSecond = (long long)64 * ldbN;
    } else {                          // b128 along n, scatter to 8 Bt rows
        b_k0 = tid >> 3;              // 0..31
        b_n0 = (tid & 7) * 8;         // n group (NT=8: also +64)
        pB = Bm + (long long)b_k0 * ldbK + (n0 + b_n0);
        bStep = 32 * ldbK;
        bSecond = 64;
    }

    us8 ra0, ra1, rb0, rb1;
    auto loadTiles = [&]() {
        ra0 = *(const us8*)pA;
        ra1 = *(const us8*)(pA + aSecond);
        rb0 = *(const us8*)pB;
        if constexpr (NT == 8) rb1 = *(const us8*)(pB + bSecond);
        pA += aStep;
        pB += bStep;
    };
    auto storeTiles = [&](int buf) {
        if constexpr (AMODE == 0) {
            *(us8*)&As[buf][a_r0][a_c0]      = ra0;
            *(us8*)&As[buf][a_r0 + 64][a_c0] = ra1;
        } else {
#pragma unroll
            for (int j = 0; j < 8; ++j) {
                As[buf][a_r0 + j][a_c0]     = ra0[j];
                As[buf][a_r0 + 8 + j][a_c0] = ra1[j];
            }
        }
        if constexpr (BMODE == 0) {
            *(us8*)&Bt[buf][b_n0][b_k0] = rb0;
            if constexpr (NT == 8) *(us8*)&Bt[buf][b_n0 + 64][b_k0] = rb1;
        } else {
#pragma unroll
            for (int j = 0; j < 8; ++j) {
                Bt[buf][b_n0 + j][b_k0] = rb0[j];
                if constexpr (NT == 8) Bt[buf][b_n0 + 64 + j][b_k0] = rb1[j];
            }
        }
    };
    auto compute = [&](int buf) {
        // load ALL fragments first -> single dscnt wait, then WMMA burst
        const int row = wave * 16 + l16;
        const us8 a_lo = *(const us8*)&As[buf][row][hi * 8];
        const us8 a_h2 = *(const us8*)&As[buf][row][16 + hi * 8];
        const v16bf fa = __builtin_bit_cast(
            v16bf, __builtin_shufflevector(a_lo, a_h2, 0, 1, 2, 3, 4, 5, 6, 7,
                                           8, 9, 10, 11, 12, 13, 14, 15));
        v16bf fb[NT];
#pragma unroll
        for (int nt = 0; nt < NT; ++nt) {
            const unsigned short* bp = &Bt[buf][nt * 16 + l16][hi * 8];
            const us8 b_lo = *(const us8*)bp;
            const us8 b_h2 = *(const us8*)(bp + 16);
            fb[nt] = __builtin_bit_cast(
                v16bf,
                __builtin_shufflevector(b_lo, b_h2, 0, 1, 2, 3, 4, 5, 6, 7, 8,
                                        9, 10, 11, 12, 13, 14, 15));
        }
#pragma unroll
        for (int nt = 0; nt < NT; ++nt)
            acc[nt] = __builtin_amdgcn_wmma_f32_16x16x32_bf16(
                false, fa, false, fb[nt], (short)0, acc[nt], false, false);
    };

    // ---- software pipeline: one barrier per k-step ----
    const int nK = K >> 5;
    loadTiles();
    storeTiles(0);
    __syncthreads();
    for (int ks = 0; ks < nK - 1; ++ks) {
        const int buf = ks & 1;
        loadTiles();                      // issue global loads for tile ks+1
        __builtin_prefetch(pA, 0, 1);     // speculative: tile ks+2
        __builtin_prefetch(pB, 0, 1);
        compute(buf);                     // WMMAs overlap the global loads
        storeTiles(buf ^ 1);              // waits loadcnt, fills other buffer
        __syncthreads();
    }
    compute((nK - 1) & 1);

    // --- epilogue: bias + scale + clamp. C/D layout: VGPR r -> rows r / r+8 ---
#pragma unroll
    for (int nt = 0; nt < NT; ++nt) {
        const int nn   = n0 + nt * 16 + l16;
        const float bv = bias ? bias[nn] : 0.0f;
#pragma unroll
        for (int r = 0; r < 8; ++r) {
            const int mm = m0 + wave * 16 + hi * 8 + r;
            float vl = (acc[nt][r] + bv) * scale;
            if (clampv > 0.0f) vl = fminf(fmaxf(vl, -clampv), clampv);
            const long long cidx =
                cbase + (long long)mm * ldcM + (long long)nn * ldcN;
            if (OUT_BF16)
                ((unsigned short*)Cv)[cidx] = f2bf(vl);
            else
                ((float*)Cv)[cidx] = vl;
        }
    }
}

// ---------------------------------------------------------------------------
// In-place softmax over the T axis of scores[BH, T, S] (bf16).
// One thread per (bh, s) column; adjacent threads read adjacent s -> coalesced.
// ---------------------------------------------------------------------------
__global__ void k_softmax_over_T(unsigned short* __restrict__ attn, int BH,
                                 int T, int S) {
    const int idx = blockIdx.x * blockDim.x + threadIdx.x;
    if (idx >= BH * S) return;
    const int bh = idx / S;
    const int s  = idx - bh * S;
    unsigned short* col = attn + (long long)bh * T * S + s;

    float m = -3.4e38f;
    for (int t = 0; t < T; ++t) {
        const float x = bf2f(col[(long long)t * S]);
        m = fmaxf(m, x);
    }
    float sum = 0.0f;
    for (int t = 0; t < T; ++t)
        sum += __expf(bf2f(col[(long long)t * S]) - m);
    const float inv = 1.0f / sum;
    for (int t = 0; t < T; ++t) {
        const float e = __expf(bf2f(col[(long long)t * S]) - m) * inv;
        col[(long long)t * S] = f2bf(e);
    }
}

// ---------------------------------------------------------------------------
// Host-side orchestration
// ---------------------------------------------------------------------------
extern "C" void kernel_launch(void* const* d_in, const int* in_sizes, int n_in,
                              void* d_out, int out_size, void* d_ws,
                              size_t ws_size, hipStream_t stream) {
    constexpr int Bb = 8, T = 2048, S = 512, VD = 512, AD = 256, E = 512,
                  H = 8, Dh = 64;

    const float* v    = (const float*)d_in[0];
    const float* a    = (const float*)d_in[1];
    const float* w_vq = (const float*)d_in[2];
    const float* b_vq = (const float*)d_in[3];
    const float* w_ak = (const float*)d_in[4];
    const float* b_ak = (const float*)d_in[5];
    const float* w_vv = (const float*)d_in[6];
    const float* b_vv = (const float*)d_in[7];
    const float* w_av = (const float*)d_in[8];
    const float* b_av = (const float*)d_in[9];
    const float* w_ov = (const float*)d_in[10];
    const float* b_ov = (const float*)d_in[11];
    const float* w_oa = (const float*)d_in[12];
    const float* b_oa = (const float*)d_in[13];
    float* outp = (float*)d_out;

    // ---- workspace carve-up (bf16 buffers), ~218 MB total ----
    char* ws = (char*)d_ws;
    size_t off = 0;
    auto carve = [&](size_t bytes) -> unsigned short* {
        void* p = ws + off;
        off += (bytes + 255) & ~(size_t)255;
        return (unsigned short*)p;
    };
    unsigned short* vbf  = carve((size_t)Bb * T * VD * 2);
    unsigned short* abf  = carve((size_t)Bb * S * AD * 2);
    unsigned short* wvq  = carve((size_t)VD * E * 2);
    unsigned short* wak  = carve((size_t)AD * E * 2);
    unsigned short* wvv  = carve((size_t)VD * E * 2);
    unsigned short* wav  = carve((size_t)AD * E * 2);
    unsigned short* wov  = carve((size_t)E * VD * 2);
    unsigned short* woa  = carve((size_t)E * AD * 2);
    unsigned short* qbf  = carve((size_t)Bb * T * E * 2);
    unsigned short* kbf  = carve((size_t)Bb * S * E * 2);
    unsigned short* vvbf = carve((size_t)Bb * T * E * 2);
    unsigned short* vabf = carve((size_t)Bb * S * E * 2);
    unsigned short* attn = carve((size_t)Bb * H * T * S * 2); // scores->attn
    unsigned short* ovbf = carve((size_t)Bb * T * E * 2);
    unsigned short* oabf = carve((size_t)Bb * S * E * 2);
    (void)ws_size; (void)n_in; (void)in_sizes; (void)out_size;

    auto cvt = [&](const float* src, unsigned short* dst, int n) {
        int blocks = (n + 1023) / 1024;
        if (blocks > 8192) blocks = 8192;
        k_f32_to_bf16<<<blocks, 256, 0, stream>>>(src, dst, n);
    };
    cvt(v, vbf, Bb * T * VD);
    cvt(a, abf, Bb * S * AD);
    cvt(w_vq, wvq, VD * E);
    cvt(w_ak, wak, AD * E);
    cvt(w_vv, wvv, VD * E);
    cvt(w_av, wav, AD * E);
    cvt(w_ov, wov, E * VD);
    cvt(w_oa, woa, E * AD);

    const dim3 blk(256);
    const float qscale = 0.125f; // D^-0.5, D=64

    // q = (v @ w_vq + b_vq) * scale        [B*T, E], bf16 out
    k_gemm_bf16_wmma<true, 0, 1, 8>
        <<<dim3(E / 128, (Bb * T) / 128, 1), blk, 0, stream>>>(
            vbf, VD, 1, 0, 0, wvq, E, 1, 0, 0, qbf, E, 1, 0, 0, b_vq, qscale,
            0.0f, VD, 1);
    // k = a @ w_ak + b_ak                  [B*S, E]
    k_gemm_bf16_wmma<true, 0, 1, 8>
        <<<dim3(E / 128, (Bb * S) / 128, 1), blk, 0, stream>>>(
            abf, AD, 1, 0, 0, wak, E, 1, 0, 0, kbf, E, 1, 0, 0, b_ak, 1.0f,
            0.0f, AD, 1);
    // val_v = v @ w_vv + b_vv              [B*T, E]
    k_gemm_bf16_wmma<true, 0, 1, 8>
        <<<dim3(E / 128, (Bb * T) / 128, 1), blk, 0, stream>>>(
            vbf, VD, 1, 0, 0, wvv, E, 1, 0, 0, vvbf, E, 1, 0, 0, b_vv, 1.0f,
            0.0f, VD, 1);
    // val_a = a @ w_av + b_av              [B*S, E]
    k_gemm_bf16_wmma<true, 0, 1, 8>
        <<<dim3(E / 128, (Bb * S) / 128, 1), blk, 0, stream>>>(
            abf, AD, 1, 0, 0, wav, E, 1, 0, 0, vabf, E, 1, 0, 0, b_av, 1.0f,
            0.0f, AD, 1);

    // scores[b,h,t,s] = clamp(q_h @ k_h^T), batched over z = b*H + h
    k_gemm_bf16_wmma<true, 0, 0, 8>
        <<<dim3(S / 128, T / 128, Bb * H), blk, 0, stream>>>(
            qbf, E, 1, (long long)T * E, Dh,    // A(t,kk) = q[b,t,h*64+kk]
            kbf, 1, E, (long long)S * E, Dh,    // B(kk,s) = k[b,s,h*64+kk]
            attn, S, 1, (long long)H * T * S, (long long)T * S, nullptr, 1.0f,
            CLAMP_VAL, Dh, H);

    // softmax over T (shared by attn_v and attn_a = attn_v^T), in place
    {
        const int ncols = Bb * H * S;
        k_softmax_over_T<<<(ncols + 255) / 256, 256, 0, stream>>>(attn, Bb * H,
                                                                  T, S);
    }

    // out_v[b,t,h,:] = attn[b,h,t,:] @ val_a[b,:,h,:]   (M=T,N=Dh,K=S)
    k_gemm_bf16_wmma<true, 0, 1, 4>
        <<<dim3(Dh / 64, T / 128, Bb * H), blk, 0, stream>>>(
            attn, S, 1, (long long)H * T * S, (long long)T * S,
            vabf, E, 1, (long long)S * E, Dh,
            ovbf, E, 1, (long long)T * E, Dh, nullptr, 1.0f, 0.0f, S, H);

    // out_a[b,s,h,:] = attn^T[b,h,s,:] @ val_v[b,:,h,:] (M=S,N=Dh,K=T)
    k_gemm_bf16_wmma<true, 1, 1, 4>
        <<<dim3(Dh / 64, S / 128, Bb * H), blk, 0, stream>>>(
            attn, 1, S, (long long)H * T * S, (long long)T * S, // A(s,t)
            vvbf, E, 1, (long long)T * E, Dh,
            oabf, E, 1, (long long)S * E, Dh, nullptr, 1.0f, 0.0f, T, H);

    // final projections -> f32 into d_out (out_v then out_a, concatenated)
    k_gemm_bf16_wmma<false, 0, 1, 8>
        <<<dim3(VD / 128, (Bb * T) / 128, 1), blk, 0, stream>>>(
            ovbf, E, 1, 0, 0, wov, VD, 1, 0, 0, outp, VD, 1, 0, 0, b_ov, 1.0f,
            0.0f, E, 1);
    k_gemm_bf16_wmma<false, 0, 1, 8>
        <<<dim3(AD / 128, (Bb * S) / 128, 1), blk, 0, stream>>>(
            oabf, E, 1, 0, 0, woa, AD, 1, 0, 0, outp + (size_t)Bb * T * VD, AD,
            1, 0, 0, b_oa, 1.0f, 0.0f, E, 1);
}
